// GCN2_23742579212601
// MI455X (gfx1250) — matrix operand order repaired and checked by reading, not compiled, not run on hardware
//
#include <hip/hip_runtime.h>
#include <math.h>

// Problem constants (from reference)
#define NN    40000
#define EE    640000
#define IN_F  256
#define HH    128
#define CC    64
#define LL    4
#define ROWS  64   // node rows per block = 4 x 16-row WMMA tiles

typedef __attribute__((ext_vector_type(16))) _Float16 v16h;
typedef __attribute__((ext_vector_type(8)))  _Float16 v8h;
typedef __attribute__((ext_vector_type(8)))  float    v8f;
typedef __attribute__((ext_vector_type(4)))  unsigned v4u;
typedef __attribute__((ext_vector_type(8)))  unsigned v8u;

#define WMMA_F16(a, b, c)                                                      \
  __builtin_amdgcn_wmma_f32_16x16x32_f16(false, (a), false, (b), (short)0,     \
                                         (c), false, false)

// ---------------------------------------------------------------------------
// Async global->LDS block-cooperative copy (GLOBAL_LOAD_ASYNC_TO_LDS_B128).
// Generic LDS pointers are {aperture_hi32, lds_offset}: low 32 bits = LDS addr.
// ---------------------------------------------------------------------------
static __device__ __forceinline__ void async_copy_g2l(const void* g, void* l,
                                                      int bytes) {
  const int t  = threadIdx.x;
  const int nt = blockDim.x;
  const unsigned lbase = (unsigned)(unsigned long long)l;
  const unsigned long long gbase = (unsigned long long)g;
  for (int off = t * 16; off < bytes; off += nt * 16) {
    const unsigned la = lbase + off;
    const unsigned long long ga = gbase + off;
    asm volatile("global_load_async_to_lds_b128 %0, %1, off"
                 :: "v"(la), "v"(ga) : "memory");
  }
}

static __device__ __forceinline__ void wait_async_and_sync() {
#if __has_builtin(__builtin_amdgcn_s_wait_asynccnt)
  __builtin_amdgcn_s_wait_asynccnt(0);
#else
  asm volatile("s_wait_asynccnt 0x0" ::: "memory");
#endif
  __syncthreads();
}

// ---------------------------------------------------------------------------
// Tensor Data Mover: 1D (single-row tile) global->LDS DMA of nhalf f16 elems.
// D# built per CDNA5 ISA: group0 (4 SGPRs) + group1 (8 SGPRs),
// VADDR2/3 = NULL (<=2D tensor). data_size=1 (2 bytes), type=2 ("image").
// ---------------------------------------------------------------------------
static __device__ __forceinline__ void tdm_load_1d(unsigned lds_addr,
                                                   const void* g,
                                                   unsigned nhalf) {
  const unsigned long long ga = (unsigned long long)g;
  v4u g0;
  g0[0] = 1u;                                              // count=1, user mode
  g0[1] = lds_addr;                                        // lds_addr[31:0]
  g0[2] = (unsigned)ga;                                    // global_addr[31:0]
  g0[3] = (unsigned)((ga >> 32) & 0x01FFFFFFu) | (2u << 30);  // addr[56:32]|type=2
  v8u g1;
  g1[0] = (1u << 16);                        // data_size=1 (2B), wg_mask=0
  g1[1] = (nhalf & 0xFFFFu) << 16;           // tensor_dim0[15:0]
  g1[2] = (nhalf >> 16) | (1u << 16);        // tensor_dim0[31:16] | tensor_dim1=1
  g1[3] = (nhalf & 0xFFFFu) << 16;           // tile_dim0
  g1[4] = 1u;                                // tile_dim1=1, tile_dim2=0
  g1[5] = nhalf;                             // tensor_dim0_stride[31:0]
  g1[6] = 0u;
  g1[7] = 0u;
  asm volatile("tensor_load_to_lds %0, %1" :: "s"(g0), "s"(g1) : "memory");
}

static __device__ __forceinline__ void wait_tensor() {
#if __has_builtin(__builtin_amdgcn_s_wait_tensorcnt)
  __builtin_amdgcn_s_wait_tensorcnt(0);
#else
  asm volatile("s_wait_tensorcnt 0x0" ::: "memory");
#endif
}

// ---------------------------------------------------------------------------
// WMMA fragment loaders (CDNA5 16x16x32 f16 layouts, wave32)
// A (16x32 MxK): lanes 0-15 -> M=lane, K={0..7,16..23}; lanes 16-31 -> K+8
// ---------------------------------------------------------------------------
static __device__ __forceinline__ v16h load_A_f32(const float* __restrict__ s,
                                                  int ld, int row0, int k0) {
  const int lane = threadIdx.x & 31;
  const int l    = lane & 15;
  const int hi   = lane >> 4;
  const float* p = s + (size_t)(row0 + l) * ld + (k0 + 8 * hi);
  v16h a;
#pragma unroll
  for (int j = 0; j < 16; ++j) {
    const int k = j + ((j >> 3) << 3);  // j<8 -> j ; j>=8 -> j+8
    a[j] = (_Float16)p[k];
  }
  return a;
}

// A fragment from f16 source: two contiguous 16B segments, no conversion
static __device__ __forceinline__ v16h load_A_f16(const _Float16* __restrict__ s,
                                                  int ld, int row0, int k0) {
  const int lane = threadIdx.x & 31;
  const int l    = lane & 15;
  const int hi   = lane >> 4;
  const _Float16* p = s + (size_t)(row0 + l) * ld + (k0 + 8 * hi);
  const v8h lo  = *(const v8h*)p;
  const v8h hi8 = *(const v8h*)(p + 16);
  v16h a;
#pragma unroll
  for (int j = 0; j < 8; ++j) {
    a[j]     = lo[j];
    a[j + 8] = hi8[j];
  }
  return a;
}

// B (32x16 KxN) from f16 N-major (transposed) panel in LDS:
// lane n=col0+(lane&15) -> column; K = k0 + 8*hi + {0..7,16..23} (matches A)
static __device__ __forceinline__ v16h load_B_lds(const _Float16* sw, int ldk,
                                                  int k0, int col0) {
  const int lane = threadIdx.x & 31;
  const int n    = col0 + (lane & 15);
  const int hi   = lane >> 4;
  const _Float16* p = sw + (size_t)n * ldk + k0 + 8 * hi;
  const v8h lo  = *(const v8h*)p;
  const v8h hi8 = *(const v8h*)(p + 16);
  v16h b;
#pragma unroll
  for (int j = 0; j < 8; ++j) {
    b[j]     = lo[j];
    b[j + 8] = hi8[j];
  }
  return b;
}

// ---------------------------------------------------------------------------
// Weight prep: src [K][N] fp32 -> dst [N][K] f16
// ---------------------------------------------------------------------------
__global__ void transpose_f16_kernel(const float* __restrict__ src,
                                     _Float16* __restrict__ dst, int K, int N) {
  const int i = blockIdx.x * blockDim.x + threadIdx.x;
  if (i < K * N) {
    const int n = i / K;
    const int k = i % K;
    dst[i] = (_Float16)src[k * N + n];
  }
}

// ---------------------------------------------------------------------------
// Degree / normalization
// ---------------------------------------------------------------------------
__global__ void deg_kernel(const int* __restrict__ src, const int* __restrict__ dst,
                           float* __restrict__ deg_out, float* __restrict__ deg_in) {
  const int e = blockIdx.x * blockDim.x + threadIdx.x;
  if (e < EE) {
    atomicAdd(&deg_out[src[e]], 1.0f);
    atomicAdd(&deg_in[dst[e]], 1.0f);
  }
}

__global__ void norm_kernel(float* __restrict__ deg_out, float* __restrict__ deg_in) {
  const int i = blockIdx.x * blockDim.x + threadIdx.x;
  if (i < NN) {
    deg_out[i] = rsqrtf(fmaxf(deg_out[i], 1.0f));
    deg_in[i]  = rsqrtf(fmaxf(deg_in[i], 1.0f));
  }
}

// ---------------------------------------------------------------------------
// fc1: v = relu(x @ fc1_w + b); h = v (fp32), h0h = (f16)(0.5*v).
// Weight panel (64KB f16) async-staged in LDS.
// ---------------------------------------------------------------------------
__global__ void __launch_bounds__(256)
fc1_kernel(const float* __restrict__ x, const _Float16* __restrict__ wT,
           const float* __restrict__ bias, float* __restrict__ h,
           _Float16* __restrict__ h0h) {
  extern __shared__ __align__(64) char smem[];
  _Float16* sW = (_Float16*)smem;  // [HH][IN_F]
  async_copy_g2l(wT, sW, HH * IN_F * (int)sizeof(_Float16));
  wait_async_and_sync();

  const int wave = threadIdx.x >> 5;
  const int col0 = wave * 16;
  const int base = blockIdx.x * ROWS;
  v8f acc[4] = {};
  for (int k0 = 0; k0 < IN_F; k0 += 32) {
    const v16h b = load_B_lds(sW, IN_F, k0, col0);
#pragma unroll
    for (int rt = 0; rt < 4; ++rt) {
      const v16h a = load_A_f32(x, IN_F, base + rt * 16, k0);
      acc[rt] = WMMA_F16(a, b, acc[rt]);
    }
  }
  const int lane = threadIdx.x & 31;
  const int n    = col0 + (lane & 15);
  const int hi   = lane >> 4;
  const float bv = bias[n];
#pragma unroll
  for (int rt = 0; rt < 4; ++rt) {
#pragma unroll
    for (int r = 0; r < 8; ++r) {
      const int m = base + rt * 16 + r + 8 * hi;
      float v = acc[rt][r] + bv;
      v = v > 0.0f ? v : 0.0f;
      const size_t idx = (size_t)m * HH + n;
      h[idx]   = v;
      h0h[idx] = (_Float16)(0.5f * v);  // ALPHA*h0, pre-scaled f16 mirror
    }
  }
}

// ---------------------------------------------------------------------------
// SpMM scatter with both norms folded:
// feat[d] += h[s] * (norm_s[s] * norm_d[d] * 0.5)      (edge x (H/4) threads)
// ---------------------------------------------------------------------------
__global__ void spmm_kernel(const float* __restrict__ h,
                            const float* __restrict__ norm_s,
                            const float* __restrict__ norm_d,
                            const int* __restrict__ src,
                            const int* __restrict__ dst,
                            float* __restrict__ feat) {
  const long long tid = (long long)blockIdx.x * blockDim.x + threadIdx.x;
  const int chunks = HH / 4;  // 32
  if (tid >= (long long)EE * chunks) return;
  const int e   = (int)(tid / chunks);
  const int cix = (int)(tid % chunks);
  const int s = src[e];
  const int d = dst[e];
  const float w = norm_s[s] * norm_d[d] * 0.5f;  // (1-ALPHA) folded
  const float4 v = ((const float4*)(h + (size_t)s * HH))[cix];
  float* o = feat + (size_t)d * HH + cix * 4;
  atomicAdd(o + 0, v.x * w);
  atomicAdd(o + 1, v.y * w);
  atomicAdd(o + 2, v.z * w);
  atomicAdd(o + 3, v.w * w);
}

// ---------------------------------------------------------------------------
// Layer: h = relu((1-b)*(feat+f0) + b*(feat@W1 + f0@W2) + bias + h)
// feat pre-scaled by spmm; f0 = h0h (pre-scaled f16). Weight panels staged in
// LDS by the Tensor Data Mover (wave 0 issues 2 descriptors, TENSORcnt wait).
// ---------------------------------------------------------------------------
__global__ void __launch_bounds__(256)
layer_kernel(const float* __restrict__ feat, const _Float16* __restrict__ h0h,
             float* __restrict__ h, const _Float16* __restrict__ W1T,
             const _Float16* __restrict__ W2T, const float* __restrict__ bias,
             float beta) {
  extern __shared__ __align__(64) char smem[];
  _Float16* sW1 = (_Float16*)smem;            // [HH][HH]
  _Float16* sW2 = sW1 + HH * HH;              // [HH][HH]
  if (threadIdx.x < 32) {                     // wave 0 drives the TDM
    tdm_load_1d((unsigned)(unsigned long long)sW1, W1T, HH * HH);
    tdm_load_1d((unsigned)(unsigned long long)sW2, W2T, HH * HH);
    wait_tensor();
  }
  __syncthreads();

  const int wave = threadIdx.x >> 5;
  const int col0 = wave * 16;
  const int base = blockIdx.x * ROWS;

  v8f acc[4] = {};
  for (int k0 = 0; k0 < HH; k0 += 32) {
    const v16h b1 = load_B_lds(sW1, HH, k0, col0);
    const v16h b2 = load_B_lds(sW2, HH, k0, col0);
#pragma unroll
    for (int rt = 0; rt < 4; ++rt) {
      const int row0 = base + rt * 16;
      const v16h a1 = load_A_f32(feat, HH, row0, k0);
      acc[rt] = WMMA_F16(a1, b1, acc[rt]);
      const v16h a2 = load_A_f16(h0h, HH, row0, k0);
      acc[rt] = WMMA_F16(a2, b2, acc[rt]);
    }
  }

  const int lane = threadIdx.x & 31;
  const int hi   = lane >> 4;
  const int n    = col0 + (lane & 15);
  const float bv  = bias[n];
  const float omb = 1.0f - beta;
#pragma unroll
  for (int rt = 0; rt < 4; ++rt) {
#pragma unroll
    for (int r = 0; r < 8; ++r) {
      const int m = base + rt * 16 + r + 8 * hi;
      const size_t idx = (size_t)m * HH + n;
      const float ft = feat[idx];
      const float f0 = (float)h0h[idx];
      float v = omb * (ft + f0) + beta * acc[rt][r] + bv + h[idx];
      h[idx] = v > 0.0f ? v : 0.0f;
    }
  }
}

// ---------------------------------------------------------------------------
// fc2: out = h @ fc2_w + fc2_b (K=128, C=64 -> 4 waves, LDS panel 16KB)
// ---------------------------------------------------------------------------
__global__ void __launch_bounds__(128)
fc2_kernel(const float* __restrict__ h, const _Float16* __restrict__ wT,
           const float* __restrict__ bias, float* __restrict__ out) {
  extern __shared__ __align__(64) char smem[];
  _Float16* sW = (_Float16*)smem;  // [CC][HH]
  async_copy_g2l(wT, sW, CC * HH * (int)sizeof(_Float16));
  wait_async_and_sync();

  const int wave = threadIdx.x >> 5;
  const int col0 = wave * 16;
  const int base = blockIdx.x * ROWS;
  v8f acc[4] = {};
  for (int k0 = 0; k0 < HH; k0 += 32) {
    const v16h b = load_B_lds(sW, HH, k0, col0);
#pragma unroll
    for (int rt = 0; rt < 4; ++rt) {
      const v16h a = load_A_f32(h, HH, base + rt * 16, k0);
      acc[rt] = WMMA_F16(a, b, acc[rt]);
    }
  }
  const int lane = threadIdx.x & 31;
  const int n    = col0 + (lane & 15);
  const int hi   = lane >> 4;
  const float bv = bias[n];
#pragma unroll
  for (int rt = 0; rt < 4; ++rt) {
#pragma unroll
    for (int r = 0; r < 8; ++r) {
      const int m = base + rt * 16 + r + 8 * hi;
      out[(size_t)m * CC + n] = acc[rt][r] + bv;
    }
  }
}

// ---------------------------------------------------------------------------
// Host launcher (graph-capture safe: kernel launches + hipMemsetAsync only)
// ---------------------------------------------------------------------------
extern "C" void kernel_launch(void* const* d_in, const int* in_sizes, int n_in,
                              void* d_out, int out_size, void* d_ws, size_t ws_size,
                              hipStream_t stream) {
  (void)in_sizes; (void)n_in; (void)out_size; (void)ws_size;
  const float* x     = (const float*)d_in[0];
  const float* fc1_w = (const float*)d_in[1];
  const float* fc1_b = (const float*)d_in[2];
  const float* W1    = (const float*)d_in[3];
  const float* W2    = (const float*)d_in[4];
  const float* bgc   = (const float*)d_in[5];
  const float* fc2_w = (const float*)d_in[6];
  const float* fc2_b = (const float*)d_in[7];
  const int*   src   = (const int*)d_in[8];
  const int*   dst   = (const int*)d_in[9];
  float* out = (float*)d_out;

  // Workspace carve-out (256B aligned)
  char* ws = (char*)d_ws;
  size_t off = 0;
  auto carveb = [&](size_t bytes) -> void* {
    void* p = ws + off;
    off += (bytes + 255) & ~(size_t)255;
    return p;
  };
  float*    deg_out = (float*)carveb(NN * sizeof(float));           // -> norm_s
  float*    deg_in  = (float*)carveb(NN * sizeof(float));           // -> norm_d
  float*    h       = (float*)carveb((size_t)NN * HH * sizeof(float));
  float*    feat    = (float*)carveb((size_t)NN * HH * sizeof(float));
  _Float16* h0h     = (_Float16*)carveb((size_t)NN * HH * 2);       // 0.5*h0 f16
  _Float16* fc1T    = (_Float16*)carveb((size_t)IN_F * HH * 2);     // [HH][IN_F]
  _Float16* W1T     = (_Float16*)carveb((size_t)LL * HH * HH * 2);  // [L][HH][HH]
  _Float16* W2T     = (_Float16*)carveb((size_t)LL * HH * HH * 2);
  _Float16* fc2T    = (_Float16*)carveb((size_t)HH * CC * 2);       // [CC][HH]

  // Weight prep: fp32 [K][N] -> f16 [N][K]
  transpose_f16_kernel<<<(IN_F * HH + 255) / 256, 256, 0, stream>>>(fc1_w, fc1T,
                                                                    IN_F, HH);
  for (int l = 0; l < LL; ++l) {
    transpose_f16_kernel<<<(HH * HH + 255) / 256, 256, 0, stream>>>(
        W1 + (size_t)l * HH * HH, W1T + (size_t)l * HH * HH, HH, HH);
    transpose_f16_kernel<<<(HH * HH + 255) / 256, 256, 0, stream>>>(
        W2 + (size_t)l * HH * HH, W2T + (size_t)l * HH * HH, HH, HH);
  }
  transpose_f16_kernel<<<(HH * CC + 255) / 256, 256, 0, stream>>>(fc2_w, fc2T,
                                                                  HH, CC);

  // Degrees -> inverse-sqrt norms (in place)
  hipMemsetAsync(deg_out, 0, NN * sizeof(float), stream);
  hipMemsetAsync(deg_in, 0, NN * sizeof(float), stream);
  deg_kernel<<<(EE + 255) / 256, 256, 0, stream>>>(src, dst, deg_out, deg_in);
  norm_kernel<<<(NN + 255) / 256, 256, 0, stream>>>(deg_out, deg_in);

  fc1_kernel<<<NN / ROWS, 256, HH * IN_F * 2, stream>>>(x, fc1T, fc1_b, h, h0h);

  for (int l = 0; l < LL; ++l) {
    const float beta = logf(1.0f / (float)(l + 1) + 1.0f);  // LAMB = 1
    hipMemsetAsync(feat, 0, (size_t)NN * HH * sizeof(float), stream);
    const long long spmm_threads = (long long)EE * (HH / 4);
    spmm_kernel<<<(int)((spmm_threads + 255) / 256), 256, 0, stream>>>(
        h, deg_out, deg_in, src, dst, feat);
    layer_kernel<<<NN / ROWS, 256, 2 * HH * HH * 2, stream>>>(
        feat, h0h, h, W1T + (size_t)l * HH * HH, W2T + (size_t)l * HH * HH,
        bgc + (size_t)l * HH, beta);
  }

  fc2_kernel<<<NN / ROWS, 128, CC * HH * 2, stream>>>(h, fc2T, fc2_b, out);
}
